// SingleHeadMDTA_35923106464231
// MI455X (gfx1250) — matrix operand order repaired and checked by reading, not compiled, not run on hardware
//
#include <hip/hip_runtime.h>
#include <cstdint>
#include <cstddef>

typedef __attribute__((ext_vector_type(16))) _Float16 v16h;
typedef __attribute__((ext_vector_type(8)))  float    v8f;
typedef __attribute__((ext_vector_type(4)))  uint32_t u32x4v;

// TDM descriptor group types (amdgpu-toolchain clang-23: 6-arg builtin)
typedef __attribute__((ext_vector_type(4))) uint32_t tdm_u4;
typedef __attribute__((ext_vector_type(8))) int      tdm_i8;
typedef __attribute__((ext_vector_type(4))) int      tdm_i4;

#if defined(__has_builtin)
#if __has_builtin(__builtin_amdgcn_tensor_load_to_lds) && \
    __has_builtin(__builtin_amdgcn_s_wait_tensorcnt)
#define USE_TDM 1
#endif
#endif

#define HW   65536   // H*W
#define Wd   256
#define Hd   256
#define C    48
#define CP   64      // channel dim padded to 64 (two K=32 WMMA steps)
#define NB   4

union F16Frag { v16h v; uint32_t u[8]; _Float16 h[16]; u32x4v q[2]; };

// ---------------------------------------------------------------------------
// Kernel 1: 1x1 pointwise conv for Q,K,V via WMMA.
//   GEMM: D[cout, pix] = sum_cin W[cout,cin] * x[cin,pix]
//   A = weights (LDS f16, cin padded to 64, ds_load_b128 fragments),
//   B = x (f32 -> f16 on load). 18 WMMA per wave (3 proj * 3 Mtiles * 2 K).
// ---------------------------------------------------------------------------
__global__ __launch_bounds__(256) void mdta_k1_pointwise(
    const float* __restrict__ x,
    const float* __restrict__ wq,
    const float* __restrict__ wk,
    const float* __restrict__ wv,
    _Float16* __restrict__ Y)   // [3][NB][C][HW] f16
{
  __shared__ _Float16 Wl[3 * C * CP];

  // Stage weights into LDS as f16, [proj][cout][64] with cin>=48 zeroed.
  for (int i = threadIdx.x; i < 3 * C * (CP / 2); i += 256) {
    int h0   = i * 2;                    // half index
    int proj = h0 / (C * CP);
    int rem  = h0 % (C * CP);
    int cout = rem / CP;
    int cin  = rem % CP;
    const float* wp = (proj == 0) ? wq : (proj == 1) ? wk : wv;
    float a = (cin     < C) ? wp[cout * C + cin]     : 0.f;
    float b = (cin + 1 < C) ? wp[cout * C + cin + 1] : 0.f;
    Wl[h0]     = (_Float16)a;
    Wl[h0 + 1] = (_Float16)b;
  }
  __syncthreads();

  const int wave    = threadIdx.x >> 5;
  const int lane    = threadIdx.x & 31;
  const int b       = blockIdx.x >> 9;           // 512 blocks per batch
  const int tile    = (blockIdx.x & 511) * 8 + wave;
  const int pixbase = tile * 16;
  const int nloc    = lane & 15;
  const int khi     = (lane < 16) ? 0 : 8;       // 16-bit A K lane-half split
  const float* xb   = x + (size_t)b * C * HW;

  // B fragments: B[k=cin, n=pix], two K-steps (cin 0..31 and 32..63 zero-pad)
  F16Frag B0, B1;
  const int pix = pixbase + nloc;
  #pragma unroll
  for (int v = 0; v < 8; ++v) {
    int k = (v < 4) ? (khi + 2 * v) : (16 + khi + 2 * (v - 4));
    B0.h[2 * v]     = (_Float16)xb[(size_t)k * HW + pix];
    B0.h[2 * v + 1] = (_Float16)xb[(size_t)(k + 1) * HW + pix];
    int k1 = 32 + k;
    float a = (k1     < C) ? xb[(size_t)k1 * HW + pix]       : 0.f;
    float c = (k1 + 1 < C) ? xb[(size_t)(k1 + 1) * HW + pix] : 0.f;
    B1.h[2 * v]     = (_Float16)a;
    B1.h[2 * v + 1] = (_Float16)c;
  }

  const int mofs = (lane < 16) ? 0 : 8;
  for (int proj = 0; proj < 3; ++proj) {
    _Float16* Yp = Y + ((size_t)proj * NB + b) * C * HW;
    #pragma unroll
    for (int mt = 0; mt < 3; ++mt) {
      const int cout = mt * 16 + nloc;
      const _Float16* wrow = &Wl[(proj * C + cout) * CP];
      F16Frag A0, A1;
      A0.q[0] = *(const u32x4v*)(wrow + khi);        // K khi..khi+7
      A0.q[1] = *(const u32x4v*)(wrow + 16 + khi);   // K 16+khi..+7
      A1.q[0] = *(const u32x4v*)(wrow + 32 + khi);
      A1.q[1] = *(const u32x4v*)(wrow + 48 + khi);
      v8f acc = {};
      acc = __builtin_amdgcn_wmma_f32_16x16x32_f16(false, A0.v, false, B0.v,
                                                   (short)0, acc, false, false);
      acc = __builtin_amdgcn_wmma_f32_16x16x32_f16(false, A1.v, false, B1.v,
                                                   (short)0, acc, false, false);
      // D: lane column n = pixel(nloc); VGPR r -> cout row r + mofs
      #pragma unroll
      for (int r = 0; r < 8; ++r) {
        int co = mt * 16 + r + mofs;
        Yp[(size_t)co * HW + pixbase + nloc] = (_Float16)acc[r];
      }
    }
  }
}

// ---------------------------------------------------------------------------
// Kernel 2: 3x3 depthwise conv (VALU, 9-tap, L2-resident Y) producing Q,K row
// tiles in LDS + V (pixel-major, padded) in global; then WMMA accumulation of
// scores[b,c,d] = sum_p K[c,p]*Q[d,p], K-dim = 32 pixels per WMMA step.
// Grid: NB*32 blocks; block processes 8 rows; wave owns a 32-pixel chunk.
// ---------------------------------------------------------------------------
__global__ __launch_bounds__(256) void mdta_k2_dw_scores(
    const _Float16* __restrict__ Y,   // [3][NB][C][HW]
    const float* __restrict__ dwq,
    const float* __restrict__ dwk,
    const float* __restrict__ dwv,
    _Float16* __restrict__ Vws,       // [NB][HW][CP]
    float* __restrict__ scores)       // [NB][C][C], pre-zeroed
{
  __shared__ _Float16 Qt[C * Wd];
  __shared__ _Float16 Kt[C * Wd];
  __shared__ float    Dw[3 * C * 9];

  for (int i = threadIdx.x; i < 3 * C * 9; i += 256) {
    int t = i / (C * 9);
    const float* dwp = (t == 0) ? dwq : (t == 1) ? dwk : dwv;
    Dw[i] = dwp[i % (C * 9)];
  }
  __syncthreads();

  const int b    = blockIdx.x >> 5;   // 32 blocks per batch
  const int blk  = blockIdx.x & 31;
  const int lane = threadIdx.x & 31;
  const int wave = threadIdx.x >> 5;
  const int nloc = lane & 15;
  const int khiA = (lane < 16) ? 0 : 8;
  const int khiB = (lane < 16) ? 0 : 16;

  v8f acc[3][3] = {};

  for (int seg = 0; seg < 8; ++seg) {
    const int h = blk * 8 + seg;

    // --- Phase A: depthwise conv for this row (Q,K -> LDS; V -> global) ---
    for (int i = threadIdx.x; i < 3 * C * Wd; i += 256) {
      int t   = i / (C * Wd);
      int rem = i % (C * Wd);
      int c   = rem / Wd;
      int w   = rem % Wd;
      const _Float16* Yp = Y + ((size_t)t * NB + b) * C * HW + (size_t)c * HW;
      const float* dw = &Dw[(t * C + c) * 9];
      float s = 0.f;
      #pragma unroll
      for (int dy = -1; dy <= 1; ++dy) {
        int hh = h + dy;
        if (hh < 0 || hh >= Hd) continue;
        #pragma unroll
        for (int dx = -1; dx <= 1; ++dx) {
          int ww = w + dx;
          if (ww < 0 || ww >= Wd) continue;
          s += (float)Yp[hh * Wd + ww] * dw[(dy + 1) * 3 + (dx + 1)];
        }
      }
      if (t == 0)      Qt[c * Wd + w] = (_Float16)s;
      else if (t == 1) Kt[c * Wd + w] = (_Float16)s;
      else Vws[((size_t)b * HW + (size_t)h * Wd + w) * CP + c] = (_Float16)s;
    }
    // zero-pad V channels 48..63 for this row (thread t -> pixel t)
    {
      int w = threadIdx.x;
      uint32_t* vz =
          (uint32_t*)&Vws[((size_t)b * HW + (size_t)h * Wd + w) * CP + C];
      #pragma unroll
      for (int j = 0; j < 8; ++j) vz[j] = 0u;
    }
    __syncthreads();

    // --- Phase B: 9 WMMA per wave over its 32-pixel chunk (b128 frags) ---
    const int chunkbase = wave * 32;
    F16Frag Bf[3];
    #pragma unroll
    for (int nt = 0; nt < 3; ++nt) {
      int d = nt * 16 + nloc;
      const _Float16* qrow = &Qt[d * Wd + chunkbase];
      Bf[nt].q[0] = *(const u32x4v*)(qrow + khiA);
      Bf[nt].q[1] = *(const u32x4v*)(qrow + 16 + khiA);
    }
    #pragma unroll
    for (int mt = 0; mt < 3; ++mt) {
      int c = mt * 16 + nloc;
      const _Float16* krow = &Kt[c * Wd + chunkbase];
      F16Frag Af;
      Af.q[0] = *(const u32x4v*)(krow + khiA);
      Af.q[1] = *(const u32x4v*)(krow + 16 + khiA);
      #pragma unroll
      for (int nt = 0; nt < 3; ++nt)
        acc[mt][nt] = __builtin_amdgcn_wmma_f32_16x16x32_f16(
            false, Af.v, false, Bf[nt].v, (short)0, acc[mt][nt], false, false);
    }
    __syncthreads();
  }

  // Flush partial scores (disjoint pixels across waves/blocks -> atomicAdd)
  const int mofs = (lane < 16) ? 0 : 8;
  float* sb = scores + (size_t)b * C * C;
  #pragma unroll
  for (int mt = 0; mt < 3; ++mt)
    #pragma unroll
    for (int nt = 0; nt < 3; ++nt)
      #pragma unroll
      for (int r = 0; r < 8; ++r) {
        int c = mt * 16 + r + mofs;
        int d = nt * 16 + nloc;
        atomicAdd(&sb[c * C + d], acc[mt][nt][r]);
      }
}

// ---------------------------------------------------------------------------
// Kernel 3: softmax over d (with 1/alpha), store transposed + padded:
//   At[b][d][c] (c padded to 64 with zeros) so K4 B-fragments are b128 loads.
// ---------------------------------------------------------------------------
__global__ void mdta_k3_softmax(const float* __restrict__ scores,
                                const float* __restrict__ alpha,
                                _Float16* __restrict__ At)  // [NB][C][CP]
{
  int t = threadIdx.x;        // 256 = NB * 64
  int b = t >> 6;
  int c = t & 63;
  float inva = 1.0f / alpha[0];
  if (c < C) {
    const float* row = scores + ((size_t)b * C + c) * C;
    float m = -1e30f;
    for (int d = 0; d < C; ++d) m = fmaxf(m, row[d] * inva);
    float e[C];
    float sum = 0.f;
    for (int d = 0; d < C; ++d) {
      e[d] = __expf(row[d] * inva - m);
      sum += e[d];
    }
    float r = 1.0f / sum;
    for (int d = 0; d < C; ++d)
      At[((size_t)b * C + d) * CP + c] = (_Float16)(e[d] * r);
  } else {
    for (int d = 0; d < C; ++d)
      At[((size_t)b * C + d) * CP + c] = (_Float16)0.f;
  }
}

// ---------------------------------------------------------------------------
// Kernel 4: out[b,d,p] = sum_c V[b,p,c] * A[b,c,d] via WMMA.
//   Attention tile (48x64 f16 = 6KB) is DMA'd into LDS once per block by the
//   Tensor Data Mover (tensor_load_to_lds + s_wait_tensorcnt), then read as
//   ds_load_b128 B-fragments. A-frag = V pixel tile (global_load_b128).
//   D staged through LDS for coalesced fp32 NCHW stores.
// ---------------------------------------------------------------------------
__global__ __launch_bounds__(256) void mdta_k4_out(
    const _Float16* __restrict__ Vws,  // [NB][HW][CP]
    const _Float16* __restrict__ At,   // [NB][C][CP]
    float* __restrict__ out)           // [NB][C][HW]
{
  __shared__ float    Obuf[8 * C * 16];
  __shared__ _Float16 AtLds[C * CP];

  const int wave     = threadIdx.x >> 5;
  const int lane     = threadIdx.x & 31;
  const int nloc     = lane & 15;
  const int b        = blockIdx.x >> 9;        // 512 blocks per batch
  const int tilebase = (blockIdx.x & 511) * 8;
  const int khiA     = (lane < 16) ? 0 : 8;
  const int khiB     = (lane < 16) ? 0 : 16;
  const int mofs     = (lane < 16) ? 0 : 8;

#ifdef USE_TDM
  // One TDM descriptor per block: copy At[b] (3072 f16, contiguous) -> LDS.
  if (threadIdx.x == 0) {
    uint64_t ga  = (uint64_t)(uintptr_t)(At + (size_t)b * C * CP);
    uint32_t lds = (uint32_t)(uintptr_t)(&AtLds[0]);   // LDS byte offset
    tdm_u4 g0;
    g0[0] = 1u;                                        // count=1 (valid D#)
    g0[1] = lds;                                       // lds_addr
    g0[2] = (uint32_t)ga;                              // global_addr[31:0]
    g0[3] = (uint32_t)((ga >> 32) & 0x01FFFFFFu) | (2u << 30);  // addr|type=2
    tdm_i8 g1;
    g1[0] = (int)(1u << 16);                 // workgroup_mask=0, data_size=2B
    g1[1] = (int)((3072u & 0xFFFFu) << 16);  // tensor_dim0 = 3072 (low 16)
    g1[2] = (int)(1u << 16);                 // tensor_dim0 hi=0, tensor_dim1=1
    g1[3] = (int)((3072u & 0xFFFFu) << 16);  // tile_dim0 = 3072
    g1[4] = 1;                               // tile_dim1 = 1, tile_dim2 = 0
    g1[5] = 3072;                            // tensor_dim0_stride
    g1[6] = 0;
    g1[7] = 0;
    tdm_i4 z4 = {};
    tdm_i8 z8 = {};
    __builtin_amdgcn_tensor_load_to_lds(g0, g1, z4, z4, z8, 0);
    __builtin_amdgcn_s_wait_tensorcnt((short)0);
  }
#else
  for (int i = threadIdx.x; i < C * CP / 2; i += 256)
    ((uint32_t*)AtLds)[i] = ((const uint32_t*)(At + (size_t)b * C * CP))[i];
#endif
  __syncthreads();

  // A fragments (V): M = pixel (lane%16), K = channel (padded 64), b128 loads
  F16Frag A0, A1;
  const _Float16* vrow =
      Vws + (size_t)b * HW * CP +
      ((size_t)(tilebase + wave) * 16 + nloc) * CP;
  A0.q[0] = *(const u32x4v*)(vrow + khiA);
  A0.q[1] = *(const u32x4v*)(vrow + 16 + khiA);
  A1.q[0] = *(const u32x4v*)(vrow + 32 + khiA);
  A1.q[1] = *(const u32x4v*)(vrow + 48 + khiA);

  #pragma unroll
  for (int nt = 0; nt < 3; ++nt) {
    int d = nt * 16 + nloc;
    const _Float16* arow = &AtLds[d * CP];
    F16Frag B0, B1;
    B0.q[0] = *(const u32x4v*)(arow + khiA);
    B0.q[1] = *(const u32x4v*)(arow + 16 + khiA);
    B1.q[0] = *(const u32x4v*)(arow + 32 + khiA);
    B1.q[1] = *(const u32x4v*)(arow + 48 + khiA);
    v8f acc = {};
    acc = __builtin_amdgcn_wmma_f32_16x16x32_f16(false, A0.v, false, B0.v,
                                                 (short)0, acc, false, false);
    acc = __builtin_amdgcn_wmma_f32_16x16x32_f16(false, A1.v, false, B1.v,
                                                 (short)0, acc, false, false);
    #pragma unroll
    for (int r = 0; r < 8; ++r)
      Obuf[wave * C * 16 + (nt * 16 + nloc) * 16 + r + mofs] = acc[r];
  }
  __syncthreads();

  float* ob = out + (size_t)b * C * HW;
  for (int i = threadIdx.x; i < 8 * C * 16; i += 256) {
    int wv  = i / (C * 16);
    int idx = i % (C * 16);
    int d   = idx >> 4;
    int px  = idx & 15;
    ob[(size_t)d * HW + (size_t)(tilebase + wv) * 16 + px] = Obuf[i];
  }
}

// ---------------------------------------------------------------------------
// Launch
// ---------------------------------------------------------------------------
extern "C" void kernel_launch(void* const* d_in, const int* in_sizes, int n_in,
                              void* d_out, int out_size, void* d_ws,
                              size_t ws_size, hipStream_t stream) {
  const float* x     = (const float*)d_in[0];
  const float* wq_pw = (const float*)d_in[1];
  const float* wq_dw = (const float*)d_in[2];
  const float* wk_pw = (const float*)d_in[3];
  const float* wk_dw = (const float*)d_in[4];
  const float* wv_pw = (const float*)d_in[5];
  const float* wv_dw = (const float*)d_in[6];
  const float* alpha = (const float*)d_in[7];
  float* out = (float*)d_out;

  // Workspace layout (bytes)
  const size_t Y_OFF  = 0;                                  // 3*NB*C*HW*2
  const size_t Y_SZ   = (size_t)3 * NB * C * HW * 2;        // 75,497,472
  const size_t V_OFF  = Y_OFF + Y_SZ;                       // NB*HW*CP*2
  const size_t V_SZ   = (size_t)NB * HW * CP * 2;           // 33,554,432
  const size_t AT_OFF = V_OFF + V_SZ;                       // NB*C*CP*2
  const size_t AT_SZ  = (size_t)NB * C * CP * 2;
  const size_t SC_OFF = AT_OFF + AT_SZ;                     // NB*C*C*4
  const size_t SC_SZ  = (size_t)NB * C * C * 4;

  char* ws = (char*)d_ws;
  _Float16* Y   = (_Float16*)(ws + Y_OFF);
  _Float16* Vws = (_Float16*)(ws + V_OFF);
  _Float16* At  = (_Float16*)(ws + AT_OFF);
  float*    sc  = (float*)(ws + SC_OFF);

  // Scores accumulate via atomics -> must start from zero every launch.
  (void)hipMemsetAsync(sc, 0, SC_SZ, stream);

  mdta_k1_pointwise<<<dim3(NB * 512), dim3(256), 0, stream>>>(
      x, wq_pw, wk_pw, wv_pw, Y);
  mdta_k2_dw_scores<<<dim3(NB * 32), dim3(256), 0, stream>>>(
      Y, wq_dw, wk_dw, wv_dw, Vws, sc);
  mdta_k3_softmax<<<dim3(1), dim3(256), 0, stream>>>(sc, alpha, At);
  mdta_k4_out<<<dim3(NB * 512), dim3(256), 0, stream>>>(Vws, At, out);
}
